// MultiHeadAttention_33122787787035
// MI455X (gfx1250) — compile-verified, matrix operand order checked by reading, not run on hardware
//
#include <hip/hip_runtime.h>

#define DMODEL 1024
#define HEADS  16
#define HD     64
#define TSEQ   2048
#define BATCH  2

typedef __bf16 v16bf __attribute__((ext_vector_type(16)));
typedef __bf16 v8bf  __attribute__((ext_vector_type(8)));
typedef float  v8f   __attribute__((ext_vector_type(8)));
typedef float  v4f   __attribute__((ext_vector_type(4)));

static __device__ inline v8f wmma_bf16(v16bf a, v16bf b, v8f c) {
  // D = A(16x32 bf16) * B(32x16 bf16) + C(16x16 f32)
  return __builtin_amdgcn_wmma_f32_16x16x32_bf16(false, a, false, b, (short)0, c,
                                                 false, false);
}

static __device__ inline v8f zero8() {
  v8f z;
#pragma unroll
  for (int i = 0; i < 8; ++i) z[i] = 0.0f;
  return z;
}

// load 8 consecutive bf16 at p (16B-aligned) into dst[off..off+7]
static __device__ inline void ld8bf(const __bf16* __restrict__ p, v16bf& dst, int off) {
  v8bf u = *(const v8bf*)p;
#pragma unroll
  for (int i = 0; i < 8; ++i) dst[off + i] = u[i];
}

// ---------------------------------------------------------------------------
// Kernel 0: elementwise fp32 -> bf16 conversion (x and the 4 weight matrices).
// 8 elements / thread: 2x global_load_b128 -> 4x v_cvt_pk_bf16_f32 -> b128 store.
// ---------------------------------------------------------------------------
__global__ __launch_bounds__(256) void cvt_f32_bf16_kernel(
    const float* __restrict__ src, __bf16* __restrict__ dst, int n8) {
  const int idx = blockIdx.x * 256 + threadIdx.x;
  if (idx >= n8) return;
  const v4f* s = (const v4f*)(src + (size_t)idx * 8);
  v4f x0 = s[0];
  v4f x1 = s[1];
  v8bf o;
#pragma unroll
  for (int i = 0; i < 4; ++i) {
    o[i]     = (__bf16)x0[i];
    o[4 + i] = (__bf16)x1[i];
  }
  *(v8bf*)(dst + (size_t)idx * 8) = o;
}

// ---------------------------------------------------------------------------
// Kernel 1: fused QKV projection.  y = x @ W^T for W in {Wq, Wk, Wv} (bf16).
// Block = 8 waves sharing one 64-col N-tile; B panel (64x32 bf16, 4KB) staged
// in LDS (double-buffered). Each wave computes a 32(M) x 64(N) tile: 8 WMMAs
// per k-step. Output bf16 in [b,h,t,d] layout.
// ---------------------------------------------------------------------------
__global__ __launch_bounds__(256) void qkv_proj_kernel(
    const __bf16* __restrict__ xb,
    const __bf16* __restrict__ Wqb, const __bf16* __restrict__ Wkb,
    const __bf16* __restrict__ Wvb,
    __bf16* __restrict__ qkv /* [3][B*H][T][HD] */) {
  __shared__ __bf16 Bs[2][64 * 32];

  const int tid  = threadIdx.x;
  const int wave = tid >> 5;
  const int lane = tid & 31;

  const int NT = 48;                     // 3*1024/64 N-tiles
  const int nt     = blockIdx.x % NT;
  const int mblock = blockIdx.x / NT;    // 0..15 (4096/256 row-blocks)
  const int wsel = nt >> 4;
  const __bf16* W = (wsel == 0) ? Wqb : (wsel == 1) ? Wkb : Wvb;
  const int nbase = (nt & 15) * 64;

  const int n    = lane & 15;
  const int half = lane >> 4;
  const int kb8  = half * 8;

  const int mt32  = mblock * 8 + wave;   // 32-row tile index (0..127)
  const int rowA0 = mt32 * 32 + n;
  const __bf16* xr0 = xb + (size_t)rowA0 * DMODEL;
  const __bf16* xr1 = xr0 + 16 * DMODEL;

  // B-panel staging coords: thread -> (col sn, 8-elem k chunk sk)
  const int sn = tid >> 2;               // 0..63
  const int sk = (tid & 3) * 8;          // 0,8,16,24
  const __bf16* wsrc = W + (size_t)(nbase + sn) * DMODEL + sk;
  __bf16* sdst0 = &Bs[0][sn * 32 + sk];
  __bf16* sdst1 = &Bs[1][sn * 32 + sk];

  v8f acc[8];                            // [sub(2)][c(4)]
#pragma unroll
  for (int i = 0; i < 8; ++i) acc[i] = zero8();

  // prologue: stage k-panel 0 into buffer 0
  *(v8bf*)sdst0 = *(const v8bf*)wsrc;
  __syncthreads();

  int buf = 0;
  for (int kk = 0; kk < DMODEL; kk += 32) {
    if (kk + 32 < DMODEL) {              // stage next panel into other buffer
      v8bf t = *(const v8bf*)(wsrc + kk + 32);
      *(v8bf*)(buf ? sdst0 : sdst1) = t;
    }

    v16bf a0, a1;
    ld8bf(xr0 + kk + kb8,      a0, 0); ld8bf(xr0 + kk + kb8 + 16, a0, 8);
    ld8bf(xr1 + kk + kb8,      a1, 0); ld8bf(xr1 + kk + kb8 + 16, a1, 8);

    const __bf16* bbase = &Bs[buf][0];
#pragma unroll
    for (int c = 0; c < 4; ++c) {
      v16bf bfrag;
      const __bf16* bp = bbase + (c * 16 + n) * 32 + kb8;
      ld8bf(bp,      bfrag, 0);
      ld8bf(bp + 16, bfrag, 8);
      acc[c]     = wmma_bf16(a0, bfrag, acc[c]);
      acc[4 + c] = wmma_bf16(a1, bfrag, acc[4 + c]);
    }
    __syncthreads();
    buf ^= 1;
  }

  // store: C layout -> [wsel][b*H+h][t][d] bf16
  const int h = nbase >> 6;
  const size_t mat = (size_t)wsel * BATCH * HEADS * TSEQ * HD;
#pragma unroll
  for (int sub = 0; sub < 2; ++sub) {
#pragma unroll
    for (int c = 0; c < 4; ++c) {
      const int d = c * 16 + n;
#pragma unroll
      for (int r = 0; r < 8; ++r) {
        const int grow = mt32 * 32 + sub * 16 + r + 8 * half;
        const int b = grow >> 11;
        const int t = grow & 2047;
        qkv[mat + ((size_t)(b * HEADS + h) * TSEQ + t) * HD + d] =
            (__bf16)acc[sub * 4 + c][r];
      }
    }
  }
}

// ---------------------------------------------------------------------------
// Kernel 2: causal flash attention per (b, h, 16-row q tile); one wave/block.
// ---------------------------------------------------------------------------
__global__ __launch_bounds__(32) void attn_kernel(
    const __bf16* __restrict__ Q, const __bf16* __restrict__ K,
    const __bf16* __restrict__ V, __bf16* __restrict__ O /* [B][T][D] */) {
  __shared__ __bf16 Pl[2][16 * 32];  // ping-pong P staging (C-layout -> A-layout)

  const int lane = threadIdx.x;
  const int bh    = blockIdx.x >> 7;    // b*H + h
  const int qtile = blockIdx.x & 127;
  const int qbase = qtile * 16;
  const int b = bh >> 4;
  const int h = bh & 15;

  const __bf16* Qh = Q + (size_t)bh * TSEQ * HD;
  const __bf16* Kh = K + (size_t)bh * TSEQ * HD;
  const __bf16* Vh = V + (size_t)bh * TSEQ * HD;

  const int n    = lane & 15;
  const int half = lane >> 4;
  const int kb8  = half * 8;

  // Q fragments (A-layout, K-dim 0..31 and 32..63)
  const __bf16* qrow = Qh + (size_t)(qbase + n) * HD;
  v16bf q0, q1;
  ld8bf(qrow + kb8,       q0, 0); ld8bf(qrow + kb8 + 16,      q0, 8);
  ld8bf(qrow + 32 + kb8,  q1, 0); ld8bf(qrow + 32 + kb8 + 16, q1, 8);

  float mrow[8], lrow[8];
  v8f o[4];
#pragma unroll
  for (int r = 0; r < 8; ++r) { mrow[r] = -1e30f; lrow[r] = 0.0f; }
#pragma unroll
  for (int c = 0; c < 4; ++c) o[c] = zero8();

  const float scale = 0.125f;  // 1/sqrt(64)
  const int qend = qbase + 15;
  int buf = 0;

  for (int j0 = 0; j0 <= qend; j0 += 32, buf ^= 1) {
    // --- scores: S[16 x 32] = Q(16x64) @ K^T(64x32), two 16x16 C tiles ---
    const __bf16* k0p = Kh + (size_t)(j0 + n) * HD;
    const __bf16* k1p = Kh + (size_t)(j0 + 16 + n) * HD;
    v16bf kb0lo, kb0hi, kb1lo, kb1hi;
    ld8bf(k0p + kb8,      kb0lo, 0); ld8bf(k0p + kb8 + 16,      kb0lo, 8);
    ld8bf(k0p + 32 + kb8, kb0hi, 0); ld8bf(k0p + 32 + kb8 + 16, kb0hi, 8);
    ld8bf(k1p + kb8,      kb1lo, 0); ld8bf(k1p + kb8 + 16,      kb1lo, 8);
    ld8bf(k1p + 32 + kb8, kb1hi, 0); ld8bf(k1p + 32 + kb8 + 16, kb1hi, 8);

    v8f s0 = zero8(), s1 = zero8();
    s0 = wmma_bf16(q0, kb0lo, s0);
    s0 = wmma_bf16(q1, kb0hi, s0);
    s1 = wmma_bf16(q0, kb1lo, s1);
    s1 = wmma_bf16(q1, kb1hi, s1);

    // --- mask + online softmax (rows live across 16 lanes of each half) ---
#pragma unroll
    for (int r = 0; r < 8; ++r) {
      const int m  = r + 8 * half;
      const int qg = qbase + m;
      float v0 = s0[r] * scale; if (j0 + n > qg)      v0 = -1e30f;
      float v1 = s1[r] * scale; if (j0 + 16 + n > qg) v1 = -1e30f;

      float mx = fmaxf(v0, v1);
#pragma unroll
      for (int d = 1; d <= 8; d <<= 1) mx = fmaxf(mx, __shfl_xor(mx, d, 32));
      const float mnew = fmaxf(mrow[r], mx);
      const float corr = __expf(mrow[r] - mnew);
      mrow[r] = mnew;

      const float p0 = __expf(v0 - mnew);
      const float p1 = __expf(v1 - mnew);
      float rs = p0 + p1;
#pragma unroll
      for (int d = 1; d <= 8; d <<= 1) rs += __shfl_xor(rs, d, 32);
      lrow[r] = lrow[r] * corr + rs;

#pragma unroll
      for (int c = 0; c < 4; ++c) o[c][r] *= corr;

      Pl[buf][m * 32 + n]      = (__bf16)p0;
      Pl[buf][m * 32 + 16 + n] = (__bf16)p1;
    }
    __syncthreads();  // single-wave workgroup: cheap; orders LDS staging

    // --- P as A-operand (16x32 bf16) ---
    v16bf pA;
    const __bf16* pr = &Pl[buf][n * 32 + kb8];
    ld8bf(pr,      pA, 0);
    ld8bf(pr + 16, pA, 8);
    __syncthreads();

    // --- O(16x64) += P(16x32) @ V(32x64): 4 WMMAs ---
#pragma unroll
    for (int c = 0; c < 4; ++c) {
      v16bf vb;
#pragma unroll
      for (int i = 0; i < 8; ++i) {
        vb[i]     = Vh[(size_t)(j0 + kb8 + i) * HD + c * 16 + n];
        vb[8 + i] = Vh[(size_t)(j0 + kb8 + 16 + i) * HD + c * 16 + n];
      }
      o[c] = wmma_bf16(pA, vb, o[c]);
    }
  }

  // epilogue: normalize and store concat-heads layout [b][t][h*64+d], bf16
#pragma unroll
  for (int r = 0; r < 8; ++r) {
    const int m = r + 8 * half;
    const int t = qbase + m;
    const float inv = 1.0f / lrow[r];
#pragma unroll
    for (int c = 0; c < 4; ++c) {
      O[(size_t)(b * TSEQ + t) * DMODEL + h * HD + c * 16 + n] =
          (__bf16)(o[c][r] * inv);
    }
  }
}

// ---------------------------------------------------------------------------
// Kernel 3: output projection.  out(f32) = attn(bf16) @ Wo^T(bf16).
// Same LDS-staged 8-wave block scheme as the QKV GEMM.
// ---------------------------------------------------------------------------
__global__ __launch_bounds__(256) void out_proj_kernel(
    const __bf16* __restrict__ A /* [B*T][D] bf16 */,
    const __bf16* __restrict__ Wob, float* __restrict__ out) {
  __shared__ __bf16 Bs[2][64 * 32];

  const int tid  = threadIdx.x;
  const int wave = tid >> 5;
  const int lane = tid & 31;

  const int NT = 16;                     // 1024/64
  const int nt     = blockIdx.x % NT;
  const int mblock = blockIdx.x / NT;    // 0..15
  const int nbase = nt * 64;

  const int n    = lane & 15;
  const int half = lane >> 4;
  const int kb8  = half * 8;

  const int mt32  = mblock * 8 + wave;
  const int rowA0 = mt32 * 32 + n;
  const __bf16* ar0 = A + (size_t)rowA0 * DMODEL;
  const __bf16* ar1 = ar0 + 16 * DMODEL;

  const int sn = tid >> 2;
  const int sk = (tid & 3) * 8;
  const __bf16* wsrc = Wob + (size_t)(nbase + sn) * DMODEL + sk;
  __bf16* sdst0 = &Bs[0][sn * 32 + sk];
  __bf16* sdst1 = &Bs[1][sn * 32 + sk];

  v8f acc[8];
#pragma unroll
  for (int i = 0; i < 8; ++i) acc[i] = zero8();

  *(v8bf*)sdst0 = *(const v8bf*)wsrc;
  __syncthreads();

  int buf = 0;
  for (int kk = 0; kk < DMODEL; kk += 32) {
    if (kk + 32 < DMODEL) {
      v8bf t = *(const v8bf*)(wsrc + kk + 32);
      *(v8bf*)(buf ? sdst0 : sdst1) = t;
    }

    v16bf a0, a1;
    ld8bf(ar0 + kk + kb8,      a0, 0); ld8bf(ar0 + kk + kb8 + 16, a0, 8);
    ld8bf(ar1 + kk + kb8,      a1, 0); ld8bf(ar1 + kk + kb8 + 16, a1, 8);

    const __bf16* bbase = &Bs[buf][0];
#pragma unroll
    for (int c = 0; c < 4; ++c) {
      v16bf bfrag;
      const __bf16* bp = bbase + (c * 16 + n) * 32 + kb8;
      ld8bf(bp,      bfrag, 0);
      ld8bf(bp + 16, bfrag, 8);
      acc[c]     = wmma_bf16(a0, bfrag, acc[c]);
      acc[4 + c] = wmma_bf16(a1, bfrag, acc[4 + c]);
    }
    __syncthreads();
    buf ^= 1;
  }

#pragma unroll
  for (int sub = 0; sub < 2; ++sub) {
#pragma unroll
    for (int c = 0; c < 4; ++c) {
#pragma unroll
      for (int r = 0; r < 8; ++r) {
        const int grow = mt32 * 32 + sub * 16 + r + 8 * half;
        out[(size_t)grow * DMODEL + nbase + c * 16 + n] = acc[sub * 4 + c][r];
      }
    }
  }
}

// ---------------------------------------------------------------------------
extern "C" void kernel_launch(void* const* d_in, const int* in_sizes, int n_in,
                              void* d_out, int out_size, void* d_ws, size_t ws_size,
                              hipStream_t stream) {
  (void)in_sizes; (void)n_in; (void)out_size; (void)ws_size;
  const float* x  = (const float*)d_in[0];
  const float* Wq = (const float*)d_in[1];
  const float* Wk = (const float*)d_in[2];
  const float* Wv = (const float*)d_in[3];
  const float* Wo = (const float*)d_in[4];
  float* out = (float*)d_out;

  const size_t XE = (size_t)BATCH * TSEQ * DMODEL;            // 4M
  const size_t WE = (size_t)DMODEL * DMODEL;                  // 1M
  const size_t QKV_ELEMS = (size_t)BATCH * HEADS * TSEQ * HD; // 4M

  __bf16* ws  = (__bf16*)d_ws;
  __bf16* xb  = ws;                         // 4M
  __bf16* Wqb = xb + XE;                    // 1M
  __bf16* Wkb = Wqb + WE;                   // 1M
  __bf16* Wvb = Wkb + WE;                   // 1M
  __bf16* Wob = Wvb + WE;                   // 1M
  __bf16* Qb  = Wob + WE;                   // 4M  [B*H][T][HD]
  __bf16* Kb  = Qb + QKV_ELEMS;             // 4M
  __bf16* Vb  = Kb + QKV_ELEMS;             // 4M
  __bf16* AO  = Vb + QKV_ELEMS;             // 4M  [B][T][D]

  // 0) one-shot fp32 -> bf16 conversion of activations + weights
  cvt_f32_bf16_kernel<<<dim3((int)(XE / 8 / 256)), dim3(256), 0, stream>>>(x,  xb,  (int)(XE / 8));
  cvt_f32_bf16_kernel<<<dim3((int)(WE / 8 / 256)), dim3(256), 0, stream>>>(Wq, Wqb, (int)(WE / 8));
  cvt_f32_bf16_kernel<<<dim3((int)(WE / 8 / 256)), dim3(256), 0, stream>>>(Wk, Wkb, (int)(WE / 8));
  cvt_f32_bf16_kernel<<<dim3((int)(WE / 8 / 256)), dim3(256), 0, stream>>>(Wv, Wvb, (int)(WE / 8));
  cvt_f32_bf16_kernel<<<dim3((int)(WE / 8 / 256)), dim3(256), 0, stream>>>(Wo, Wob, (int)(WE / 8));

  // 1) fused QKV projection: 16 row-blocks x 48 N-tiles
  qkv_proj_kernel<<<dim3(16 * 48), dim3(256), 0, stream>>>(xb, Wqb, Wkb, Wvb, Qb);

  // 2) flash attention: B*H * (T/16) q-tiles, one wave each
  attn_kernel<<<dim3(BATCH * HEADS * (TSEQ / 16)), dim3(32), 0, stream>>>(Qb, Kb, Vb, AO);

  // 3) output projection: 16 row-blocks x 16 N-tiles
  out_proj_kernel<<<dim3(16 * 16), dim3(256), 0, stream>>>(AO, Wob, out);
}
